// SparseAutoencoder_90752658964571
// MI455X (gfx1250) — compile-verified, matrix operand order checked by reading, not run on hardware
//
#include <hip/hip_runtime.h>
#include <hip/hip_bf16.h>

#define DIM   1024
#define NROW  4096
#define NCODE 32768
#define TOPK  32
#define EPSV  1e-5f

typedef __bf16 v16bf __attribute__((ext_vector_type(16)));
typedef __bf16 v8bf  __attribute__((ext_vector_type(8)));
typedef float  v8f   __attribute__((ext_vector_type(8)));

// ---------- helpers ----------
__device__ __forceinline__ unsigned short f2bf(float f) {
    unsigned u = __float_as_uint(f);
    unsigned rnd = 0x7FFFu + ((u >> 16) & 1u);   // round-to-nearest-even
    return (unsigned short)((u + rnd) >> 16);
}

__device__ __forceinline__ unsigned f2key(float f) {
    unsigned u = __float_as_uint(f);
    return (u & 0x80000000u) ? ~u : (u | 0x80000000u);  // monotone order-preserving
}

// Build v16bf in the CDNA5 16-bit A/B register layout: this lane holds
// K = {base..base+7} in elems 0..7 and K = {base+16..base+23} in elems 8..15.
__device__ __forceinline__ v16bf load_ab_tile(const __bf16* __restrict__ p) {
    const v8bf lo = *reinterpret_cast<const v8bf*>(p);
    const v8bf hi = *reinterpret_cast<const v8bf*>(p + 16);
    v16bf r;
#pragma unroll
    for (int i = 0; i < 8; ++i) { r[i] = lo[i]; r[8 + i] = hi[i]; }
    return r;
}

__device__ __forceinline__ v8f wmma_bf16(v16bf a, v16bf b, v8f c) {
    return __builtin_amdgcn_wmma_f32_16x16x32_bf16(false, a, false, b, (short)0, c,
                                                   false, false);
}

// ---------- kernel 1: row stats + bf16 normalized x ----------
__global__ __launch_bounds__(256) void row_stats_kernel(
    const float* __restrict__ x, unsigned short* __restrict__ xn_bf,
    float* __restrict__ mu_out, float* __restrict__ sd_out)
{
    __shared__ float red[256];
    const int row = blockIdx.x, tid = threadIdx.x;
    const float* xr = x + (size_t)row * DIM;
    const float4 v = *reinterpret_cast<const float4*>(xr + tid * 4);

    red[tid] = v.x + v.y + v.z + v.w;
    __syncthreads();
    for (int s = 128; s > 0; s >>= 1) {
        if (tid < s) red[tid] += red[tid + s];
        __syncthreads();
    }
    const float m = red[0] * (1.0f / DIM);
    __syncthreads();

    const float cx = v.x - m, cy = v.y - m, cz = v.z - m, cw = v.w - m;
    red[tid] = cx * cx + cy * cy + cz * cz + cw * cw;
    __syncthreads();
    for (int s = 128; s > 0; s >>= 1) {
        if (tid < s) red[tid] += red[tid + s];
        __syncthreads();
    }
    const float sd  = sqrtf(red[0] * (1.0f / (DIM - 1)));   // ddof = 1
    const float inv = 1.0f / (sd + EPSV);

    unsigned short* o = xn_bf + (size_t)row * DIM + tid * 4;
    o[0] = f2bf(cx * inv); o[1] = f2bf(cy * inv);
    o[2] = f2bf(cz * inv); o[3] = f2bf(cw * inv);
    if (tid == 0) { mu_out[row] = m; sd_out[row] = sd; }
}

// ---------- kernel 2: W row norms + bf16 normalized W ----------
__global__ __launch_bounds__(256) void w_norm_kernel(
    const float* __restrict__ W, unsigned short* __restrict__ wn_bf,
    float* __restrict__ invn_out)
{
    __shared__ float red[256];
    const int row = blockIdx.x, tid = threadIdx.x;
    const float* wr = W + (size_t)row * DIM;
    const float4 v = *reinterpret_cast<const float4*>(wr + tid * 4);
    red[tid] = v.x * v.x + v.y * v.y + v.z * v.z + v.w * v.w;
    __syncthreads();
    for (int s = 128; s > 0; s >>= 1) {
        if (tid < s) red[tid] += red[tid + s];
        __syncthreads();
    }
    const float inv = rsqrtf(red[0]);
    unsigned short* o = wn_bf + (size_t)row * DIM + tid * 4;
    o[0] = f2bf(v.x * inv); o[1] = f2bf(v.y * inv);
    o[2] = f2bf(v.z * inv); o[3] = f2bf(v.w * inv);
    if (tid == 0) invn_out[row] = inv;
}

// ---------- kernel 3: encode GEMM via v_wmma_f32_16x16x32_bf16 ----------
// Wave computes a 16x64 latent tile (4 accumulators), software-pipelined:
// next K-step's A/B fragments are fetched while current step's WMMAs run,
// so the compiler emits partial s_wait_loadcnt instead of wait-0 per WMMA.
// grid = (NCODE/512, NROW/16), 8 waves/WG.
__global__ __launch_bounds__(256) void encode_wmma_kernel(
    const unsigned short* __restrict__ xn_u16,
    const unsigned short* __restrict__ wn_u16,
    float* __restrict__ latents)
{
    const __bf16* xn = reinterpret_cast<const __bf16*>(xn_u16);
    const __bf16* wn = reinterpret_cast<const __bf16*>(wn_u16);

    const int lane    = threadIdx.x & 31;
    const int wave    = threadIdx.x >> 5;
    const int l15     = lane & 15;
    const int half    = lane >> 4;                 // 0: K 0..7/16..23, 1: K 8..15/24..31
    const int rowBase = blockIdx.y * 16;
    const int colBase = (blockIdx.x * 8 + wave) * 64;

    const __bf16* aP  = xn + (size_t)(rowBase + l15) * DIM + half * 8;
    const __bf16* bP0 = wn + (size_t)(colBase +  0 + l15) * DIM + half * 8;
    const __bf16* bP1 = wn + (size_t)(colBase + 16 + l15) * DIM + half * 8;
    const __bf16* bP2 = wn + (size_t)(colBase + 32 + l15) * DIM + half * 8;
    const __bf16* bP3 = wn + (size_t)(colBase + 48 + l15) * DIM + half * 8;

    v8f acc0 = {}, acc1 = {}, acc2 = {}, acc3 = {};

    // prologue: fetch K-step 0
    v16bf a  = load_ab_tile(aP);
    v16bf b0 = load_ab_tile(bP0);
    v16bf b1 = load_ab_tile(bP1);
    v16bf b2 = load_ab_tile(bP2);
    v16bf b3 = load_ab_tile(bP3);

    for (int k = 32; k < DIM; k += 32) {
        // issue next step's loads before this step's WMMAs
        const v16bf an  = load_ab_tile(aP  + k);
        const v16bf bn0 = load_ab_tile(bP0 + k);
        const v16bf bn1 = load_ab_tile(bP1 + k);
        const v16bf bn2 = load_ab_tile(bP2 + k);
        const v16bf bn3 = load_ab_tile(bP3 + k);

        acc0 = wmma_bf16(a, b0, acc0);
        acc1 = wmma_bf16(a, b1, acc1);
        acc2 = wmma_bf16(a, b2, acc2);
        acc3 = wmma_bf16(a, b3, acc3);

        a = an; b0 = bn0; b1 = bn1; b2 = bn2; b3 = bn3;
    }
    acc0 = wmma_bf16(a, b0, acc0);
    acc1 = wmma_bf16(a, b1, acc1);
    acc2 = wmma_bf16(a, b2, acc2);
    acc3 = wmma_bf16(a, b3, acc3);

    // C/D layout: VGPR r holds M = r (lanes 0-15) / r+8 (lanes 16-31), N = lane&15
    const int outRow0 = rowBase + half * 8;
#pragma unroll
    for (int r = 0; r < 8; ++r) {
        float* o = latents + (size_t)(outRow0 + r) * NCODE + colBase + l15;
        o[0]  = acc0[r];
        o[16] = acc1[r];
        o[32] = acc2[r];
        o[48] = acc3[r];
    }
}

// ---------- kernel 4: exact top-32 per row, 3-pass wide-digit radix select ----------
// Digits: bits[31:21] (2048 buckets), bits[20:10] (2048), bits[9:0] (1024).
// 3 histogram passes + 1 merged collection pass = 4 sweeps of the row.
__global__ __launch_bounds__(256) void topk_radix_kernel(
    const float* __restrict__ latents,
    int* __restrict__ topk_idx, float* __restrict__ topk_val)
{
    __shared__ unsigned hist[2048];
    __shared__ unsigned segsum[256];
    __shared__ unsigned s_prefix;
    __shared__ int s_k;
    __shared__ int s_cnt, s_tie;
    __shared__ int   s_idx[TOPK];
    __shared__ float s_val[TOPK];
    __shared__ int   t_idx[TOPK];
    __shared__ float t_val[TOPK];

    const int row = blockIdx.x, tid = threadIdx.x;
    const float* rowp = latents + (size_t)row * NCODE;

    if (tid == 0) { s_prefix = 0u; s_k = TOPK; }
    __syncthreads();

    const int shifts[3] = { 21, 10, 0 };
    const int nbuck [3] = { 2048, 2048, 1024 };

#pragma unroll 1
    for (int pass = 0; pass < 3; ++pass) {
        const int shift = shifts[pass];
        const int nb    = nbuck[pass];
        const unsigned digmask = (unsigned)(nb - 1);
        for (int j = tid; j < nb; j += 256) hist[j] = 0u;
        __syncthreads();
        const unsigned prefix = s_prefix;
        const int      chkShift = shift + (pass == 2 ? 10 : 11);
        for (int i = tid; i < NCODE; i += 256) {
            const unsigned key = f2key(rowp[i]);
            if (pass == 0 || (key >> chkShift) == (prefix >> chkShift))
                atomicAdd(&hist[(key >> shift) & digmask], 1u);
        }
        __syncthreads();
        // parallel per-segment partial sums (8 buckets per thread)
        const int nseg = nb >> 3;
        unsigned loc = 0;
        if (tid < nseg) {
#pragma unroll
            for (int j = 0; j < 8; ++j) loc += hist[tid * 8 + j];
            segsum[tid] = loc;
        }
        __syncthreads();
        if (tid == 0) {
            int need = s_k;
            unsigned above = 0;
            int sg = nseg - 1;
            for (; sg > 0; --sg) {
                const unsigned c = segsum[sg];
                if ((int)(above + c) >= need) break;
                above += c;
            }
            int b = sg * 8 + 7;
            for (; b > sg * 8; --b) {
                const unsigned c = hist[b];
                if ((int)(above + c) >= need) break;
                above += c;
            }
            s_prefix = prefix | ((unsigned)b << shift);
            s_k = need - (int)above;
        }
        __syncthreads();
    }

    const unsigned T = s_prefix;   // exact key of the 32nd-largest element
    if (tid == 0) { s_cnt = 0; s_tie = 0; }
    __syncthreads();

    // single merged collection pass: strictly-greater + ties
    for (int i = tid; i < NCODE; i += 256) {
        const float v = rowp[i];
        const unsigned key = f2key(v);
        if (key > T) {
            const int p = atomicAdd(&s_cnt, 1);
            if (p < TOPK) { s_idx[p] = i; s_val[p] = v; }
        } else if (key == T) {
            const int p = atomicAdd(&s_tie, 1);
            if (p < TOPK) { t_idx[p] = i; t_val[p] = v; }
        }
    }
    __syncthreads();
    const int ngt = s_cnt < TOPK ? s_cnt : TOPK;
    if (tid < TOPK) {
        int   oi; float ov;
        if (tid < ngt) { oi = s_idx[tid];       ov = s_val[tid]; }
        else           { oi = t_idx[tid - ngt]; ov = t_val[tid - ngt]; }
        topk_idx[(size_t)row * TOPK + tid] = oi;
        topk_val[(size_t)row * TOPK + tid] = ov;
    }
}

// ---------- kernel 5: sparse decode + denorm ----------
__global__ __launch_bounds__(256) void decode_kernel(
    const float* __restrict__ W, const float* __restrict__ invn,
    const int* __restrict__ topk_idx, const float* __restrict__ topk_val,
    const float* __restrict__ mu, const float* __restrict__ sd,
    float* __restrict__ xhat)
{
    __shared__ int   s_idx[TOPK];
    __shared__ float s_s[TOPK];
    const int row = blockIdx.x, tid = threadIdx.x;
    if (tid < TOPK) {
        const int c = topk_idx[(size_t)row * TOPK + tid];
        s_idx[tid] = c;
        s_s[tid]   = topk_val[(size_t)row * TOPK + tid] * invn[c];
    }
    __syncthreads();

    const int d = tid * 4;
    float4 acc = {0.f, 0.f, 0.f, 0.f};
#pragma unroll 8
    for (int j = 0; j < TOPK; ++j) {
        const float4 w = *reinterpret_cast<const float4*>(W + (size_t)s_idx[j] * DIM + d);
        const float s = s_s[j];
        acc.x += s * w.x; acc.y += s * w.y; acc.z += s * w.z; acc.w += s * w.w;
    }
    const float sdv = sd[row], m = mu[row];
    float4 o = { acc.x * sdv + m, acc.y * sdv + m, acc.z * sdv + m, acc.w * sdv + m };
    *reinterpret_cast<float4*>(xhat + (size_t)row * DIM + d) = o;
}

// ---------- launcher ----------
extern "C" void kernel_launch(void* const* d_in, const int* in_sizes, int n_in,
                              void* d_out, int out_size, void* d_ws, size_t ws_size,
                              hipStream_t stream) {
    (void)in_sizes; (void)n_in; (void)out_size; (void)ws_size;
    const float* x = (const float*)d_in[0];
    const float* W = (const float*)d_in[1];

    float* out     = (float*)d_out;
    float* xhat    = out;                                 // 4096*1024
    float* latents = out + (size_t)NROW * DIM;            // 4096*32768

    char* p = (char*)d_ws;
    unsigned short* xn_bf = (unsigned short*)p; p += (size_t)NROW  * DIM * 2;  // 8 MB
    unsigned short* wn_bf = (unsigned short*)p; p += (size_t)NCODE * DIM * 2;  // 64 MB
    float* mu   = (float*)p; p += (size_t)NROW  * 4;
    float* sd   = (float*)p; p += (size_t)NROW  * 4;
    float* invn = (float*)p; p += (size_t)NCODE * 4;
    int*   tidx = (int*)p;   p += (size_t)NROW * TOPK * 4;
    float* tval = (float*)p; p += (size_t)NROW * TOPK * 4;

    row_stats_kernel<<<NROW, 256, 0, stream>>>(x, xn_bf, mu, sd);
    w_norm_kernel<<<NCODE, 256, 0, stream>>>(W, wn_bf, invn);

    dim3 grid(NCODE / 512, NROW / 16);
    encode_wmma_kernel<<<grid, 256, 0, stream>>>(xn_bf, wn_bf, latents);

    topk_radix_kernel<<<NROW, 256, 0, stream>>>(latents, tidx, tval);
    decode_kernel<<<NROW, 256, 0, stream>>>(W, invn, tidx, tval, mu, sd, xhat);
}
